// Qwen2MoELayer_86242943303729
// MI455X (gfx1250) — matrix operand order repaired and checked by reading, not compiled, not run on hardware
//
#include <hip/hip_runtime.h>
#include <hip/hip_bf16.h>
#include <stdint.h>
#include <stddef.h>

#define T_TOK 4096
#define H_DIM 2048
#define E_NUM 8
#define I_DIM 1408
#define TOPK  2

typedef __bf16 bf16;
typedef __attribute__((ext_vector_type(16))) __bf16 v16bf;
typedef __attribute__((ext_vector_type(8)))  __bf16 bf16x8;
typedef __attribute__((ext_vector_type(4)))  __bf16 bf16x4;
typedef __attribute__((ext_vector_type(8)))  float  v8f;
typedef __attribute__((ext_vector_type(4)))  float  f32x4;

#define BM   128         /* block rows (4 waves x 32)                  */
#define BN   128         /* block cols (2 waves x 64)                  */
#define KB   32          /* K-slice per stage (1 WMMA K-step)          */
#define LDK  (KB + 8)    /* padded LDS row stride (bf16 elems)         */

__device__ __forceinline__ v8f v8f_zero() {
  v8f z;
#pragma unroll
  for (int i = 0; i < 8; ++i) z[i] = 0.0f;
  return z;
}

__device__ __forceinline__ v16bf cat16(bf16x8 lo, bf16x8 hi) {
  return __builtin_shufflevector(lo, hi, 0,1,2,3,4,5,6,7,8,9,10,11,12,13,14,15);
}

// ---- CDNA5 async global->LDS copy (ASYNCcnt-tracked, no VGPR round-trip) ----
__device__ __forceinline__ void async_ld_b128(void* lds_ptr, const void* gptr) {
  uint32_t l = (uint32_t)(uintptr_t)lds_ptr;   // low 32 bits of generic = LDS offset
  uint64_t g = (uint64_t)(uintptr_t)gptr;
  asm volatile("global_load_async_to_lds_b128 %0, %1, off"
               :: "v"(l), "v"(g) : "memory");
}
__device__ __forceinline__ void wait_async0() {
  asm volatile("s_wait_asynccnt 0" ::: "memory");
}

// ---------------- fp32 -> bf16 (layout-preserving, for x) ----------------
__global__ void cvt_f32_bf16(const float* __restrict__ src, bf16* __restrict__ dst, int n4) {
  int i = blockIdx.x * blockDim.x + threadIdx.x;
  int stride = gridDim.x * blockDim.x;
  for (; i < n4; i += stride) {
    f32x4 v = ((const f32x4*)src)[i];
    bf16x4 o;
#pragma unroll
    for (int j = 0; j < 4; ++j) o[j] = (bf16)v[j];
    ((bf16x4*)dst)[i] = o;
  }
}

// ---------------- fp32 (Kd x Nd) -> bf16 transposed (Nd x Kd), per expert ----------------
__global__ void transpose_cvt(const float* __restrict__ src, bf16* __restrict__ dst,
                              int Kd, int Nd) {
  __shared__ bf16 sT[64 * 72];
  const size_t base = (size_t)blockIdx.z * Kd * Nd;
  const int n0 = blockIdx.x * 64;
  const int k0 = blockIdx.y * 64;
  const int tid = threadIdx.x;
#pragma unroll
  for (int g = 0; g < 4; ++g) {             // read 64x64 fp32 tile, coalesced along N
    int G = tid + g * 256;
    int k = G >> 4;
    int nq = (G & 15) * 4;
    f32x4 v = *(const f32x4*)(src + base + (size_t)(k0 + k) * Nd + (n0 + nq));
#pragma unroll
    for (int ii = 0; ii < 4; ++ii) sT[(nq + ii) * 72 + k] = (bf16)v[ii];
  }
  __syncthreads();
#pragma unroll
  for (int g = 0; g < 2; ++g) {             // write transposed bf16, coalesced along K
    int G = tid + g * 256;
    int n = G >> 3;
    int kq = (G & 7) * 8;
    bf16x8 v = *(const bf16x8*)&sT[n * 72 + kq];
    *(bf16x8*)(dst + base + (size_t)(n0 + n) * Kd + (k0 + kq)) = v;
  }
}

// ---------------- router: logits -> softmax -> top-2 ----------------
__global__ void router_kernel(const float* __restrict__ x, const float* __restrict__ rw,
                              int* __restrict__ topk_idx, float* __restrict__ topk_w,
                              int* __restrict__ counts) {
  __shared__ float red[128 * E_NUM];
  const int t = blockIdx.x;
  const int tid = threadIdx.x;
  float acc[E_NUM];
#pragma unroll
  for (int e = 0; e < E_NUM; ++e) acc[e] = 0.0f;
  const float* xr = x + (size_t)t * H_DIM;
  for (int j = tid; j < H_DIM; j += 128) {
    float xv = xr[j];
    const float* w = rw + (size_t)j * E_NUM;
#pragma unroll
    for (int e = 0; e < E_NUM; ++e) acc[e] += xv * w[e];
  }
#pragma unroll
  for (int e = 0; e < E_NUM; ++e) red[tid * E_NUM + e] = acc[e];
  __syncthreads();
  for (int s = 64; s > 0; s >>= 1) {
    if (tid < s) {
#pragma unroll
      for (int e = 0; e < E_NUM; ++e) red[tid * E_NUM + e] += red[(tid + s) * E_NUM + e];
    }
    __syncthreads();
  }
  if (tid == 0) {
    float p[E_NUM];
    float mx = -3.0e38f;
#pragma unroll
    for (int e = 0; e < E_NUM; ++e) { p[e] = red[e]; mx = p[e] > mx ? p[e] : mx; }
    float sum = 0.0f;
#pragma unroll
    for (int e = 0; e < E_NUM; ++e) { p[e] = __expf(p[e] - mx); sum += p[e]; }
    float inv = __builtin_amdgcn_rcpf(sum);
#pragma unroll
    for (int e = 0; e < E_NUM; ++e) p[e] *= inv;
    int i1 = 0; float w1 = -1.0f;
#pragma unroll
    for (int e = 0; e < E_NUM; ++e) if (p[e] > w1) { w1 = p[e]; i1 = e; }
    int i2 = -1; float w2 = -1.0f;
#pragma unroll
    for (int e = 0; e < E_NUM; ++e) if (e != i1 && p[e] > w2) { w2 = p[e]; i2 = e; }
    topk_idx[t * TOPK + 0] = i1;
    topk_idx[t * TOPK + 1] = i2;
    topk_w[t * TOPK + 0] = w1;
    topk_w[t * TOPK + 1] = w2;
    atomicAdd(&counts[i1], 1);
    atomicAdd(&counts[i2], 1);
  }
}

// ---------------- exclusive prefix over expert counts ----------------
__global__ void prefix_kernel(const int* __restrict__ counts, int* __restrict__ offsets,
                              int* __restrict__ fill) {
  if (blockIdx.x == 0 && threadIdx.x == 0) {
    int s = 0;
    for (int e = 0; e < E_NUM; ++e) { offsets[e] = s; s += counts[e]; fill[e] = 0; }
    offsets[E_NUM] = s;
  }
}

// ---------------- scatter (token,k) into per-expert lists ----------------
__global__ void scatter_kernel(const int* __restrict__ topk_idx, const float* __restrict__ topk_w,
                               const int* __restrict__ offsets, int* __restrict__ fill,
                               int* __restrict__ perm_tok, float* __restrict__ perm_w) {
  int i = blockIdx.x * blockDim.x + threadIdx.x;
  if (i >= T_TOK * TOPK) return;
  int e = topk_idx[i];
  int slot = offsets[e] + atomicAdd(&fill[e], 1);
  perm_tok[slot] = i / TOPK;
  perm_w[slot] = topk_w[i];
}

// ---------------- GEMM1: h = silu(X Wg) * (X Wu); Wt is (2I x H) bf16 ----------------
__launch_bounds__(256)
__global__ void moe_gemm1(const bf16* __restrict__ xb,
                          const bf16* __restrict__ wgt,
                          const int*  __restrict__ perm_tok,
                          const int*  __restrict__ offsets,
                          bf16* __restrict__ hbuf) {
  const int e     = blockIdx.z;
  const int off_e = offsets[e];
  const int cnt   = offsets[e + 1] - off_e;
  const int mbase = blockIdx.x * BM;
  if (mbase >= cnt) return;
  const int nbase = blockIdx.y * BN;            // within I

  __shared__ bf16 sA [2 * BM * LDK];            // double-buffered: 3 x 20 KB
  __shared__ bf16 sBg[2 * BN * LDK];
  __shared__ bf16 sBu[2 * BN * LDK];

  const int tid  = threadIdx.x;
  const int lane = tid & 31;
  const int wave = tid >> 5;
  const int wm   = wave & 3;                    // 4 waves along M
  const int wn   = wave >> 2;                   // 2 waves along N
  const int q    = lane >> 4;
  const int ln   = lane & 15;

  // staging maps (K-run contiguous; rows clamped -> branch-free gather)
  const int kq = (tid & 3) * 8;                 // 0..24
  const int rb = tid >> 2;                      // 0..63
  const bf16* aptr[2];
#pragma unroll
  for (int g = 0; g < 2; ++g) {
    int r = mbase + rb + g * 64;
    if (r >= cnt) r = cnt - 1;                  // extra rows discarded at epilogue
    aptr[g] = xb + (size_t)perm_tok[off_e + r] * H_DIM;
  }
  const bf16* wg_gate = wgt + (size_t)e * (2 * I_DIM) * H_DIM + (size_t)nbase * H_DIM;
  const bf16* wg_up   = wg_gate + (size_t)I_DIM * H_DIM;

  v8f accg[2][4], accu[2][4];
#pragma unroll
  for (int mi = 0; mi < 2; ++mi)
#pragma unroll
    for (int ni = 0; ni < 4; ++ni) { accg[mi][ni] = v8f_zero(); accu[mi][ni] = v8f_zero(); }

  auto issue = [&](int kb, int buf) {
    const int ab = buf * (BM * LDK);
#pragma unroll
    for (int g = 0; g < 2; ++g)                 // A tile 128x32 (gathered rows)
      async_ld_b128(&sA[ab + (rb + g * 64) * LDK + kq], aptr[g] + kb + kq);
#pragma unroll
    for (int g = 0; g < 2; ++g) {               // B tiles 128x32 (gate + up)
      int n = rb + g * 64;
      async_ld_b128(&sBg[ab + n * LDK + kq], wg_gate + (size_t)n * H_DIM + kb + kq);
      async_ld_b128(&sBu[ab + n * LDK + kq], wg_up   + (size_t)n * H_DIM + kb + kq);
    }
  };

  issue(0, 0);
  wait_async0();
  __syncthreads();

  const int S = H_DIM / KB;                     // 64 stages
  for (int s = 0; s < S; ++s) {
    const int cur = s & 1;
    if (s + 1 < S) issue((s + 1) * KB, cur ^ 1);   // DMA next stage during compute

    const bf16* A  = sA  + cur * (BM * LDK);
    const bf16* Bg = sBg + cur * (BN * LDK);
    const bf16* Bu = sBu + cur * (BN * LDK);
    v16bf af[2], bg[4], bu[4];
#pragma unroll
    for (int mi = 0; mi < 2; ++mi) {
      int m = wm * 32 + mi * 16 + ln;
      af[mi] = cat16(*(const bf16x8*)&A[m * LDK + q * 8],
                     *(const bf16x8*)&A[m * LDK + 16 + q * 8]);
    }
#pragma unroll
    for (int ni = 0; ni < 4; ++ni) {
      int n = wn * 64 + ni * 16 + ln;
      bg[ni] = cat16(*(const bf16x8*)&Bg[n * LDK + q * 8],
                     *(const bf16x8*)&Bg[n * LDK + 16 + q * 8]);
      bu[ni] = cat16(*(const bf16x8*)&Bu[n * LDK + q * 8],
                     *(const bf16x8*)&Bu[n * LDK + 16 + q * 8]);
    }
#pragma unroll
    for (int mi = 0; mi < 2; ++mi)
#pragma unroll
      for (int ni = 0; ni < 4; ++ni) {
        accg[mi][ni] = __builtin_amdgcn_wmma_f32_16x16x32_bf16(
            false, af[mi], false, bg[ni], (short)0, accg[mi][ni], false, false);
        accu[mi][ni] = __builtin_amdgcn_wmma_f32_16x16x32_bf16(
            false, af[mi], false, bu[ni], (short)0, accu[mi][ni], false, false);
      }

    if (s + 1 < S) {
      wait_async0();                            // next buffer filled (loads retire in order)
      __syncthreads();                          // everyone done reading current buffer
    }
  }

  // epilogue: fused SiLU(gate)*up -> bf16 h
#pragma unroll
  for (int mi = 0; mi < 2; ++mi)
#pragma unroll
    for (int ni = 0; ni < 4; ++ni) {
      int col = nbase + wn * 64 + ni * 16 + ln;
#pragma unroll
      for (int r = 0; r < 8; ++r) {
        int gr = mbase + wm * 32 + mi * 16 + r + q * 8;
        if (gr < cnt) {
          float gv = accg[mi][ni][r];
          float uv = accu[mi][ni][r];
          float sv = gv * __builtin_amdgcn_rcpf(1.0f + __expf(-gv));
          hbuf[(size_t)(off_e + gr) * I_DIM + col] = (bf16)(sv * uv);
        }
      }
    }
}

// ---------------- GEMM2: out += w_tok * (h @ Wd); Wdt is (H x I) bf16 ----------------
__launch_bounds__(256)
__global__ void moe_gemm2(const bf16* __restrict__ hbuf,
                          const bf16* __restrict__ wdt,
                          const int*  __restrict__ perm_tok,
                          const float* __restrict__ perm_w,
                          const int*  __restrict__ offsets,
                          float* __restrict__ out) {
  const int e     = blockIdx.z;
  const int off_e = offsets[e];
  const int cnt   = offsets[e + 1] - off_e;
  const int mbase = blockIdx.x * BM;
  if (mbase >= cnt) return;
  const int nbase = blockIdx.y * BN;            // within H

  __shared__ bf16 sA[2 * BM * LDK];
  __shared__ bf16 sB[2 * BN * LDK];

  const int tid  = threadIdx.x;
  const int lane = tid & 31;
  const int wave = tid >> 5;
  const int wm   = wave & 3;
  const int wn   = wave >> 2;
  const int q    = lane >> 4;
  const int ln   = lane & 15;

  const int kq = (tid & 3) * 8;
  const int rb = tid >> 2;
  const bf16* aptr[2];
#pragma unroll
  for (int g = 0; g < 2; ++g) {
    int r = mbase + rb + g * 64;
    if (r >= cnt) r = cnt - 1;
    aptr[g] = hbuf + (size_t)(off_e + r) * I_DIM;
  }
  const bf16* wd = wdt + (size_t)e * H_DIM * I_DIM + (size_t)nbase * I_DIM;

  v8f acc[2][4];
#pragma unroll
  for (int mi = 0; mi < 2; ++mi)
#pragma unroll
    for (int ni = 0; ni < 4; ++ni) acc[mi][ni] = v8f_zero();

  auto issue = [&](int kb, int buf) {
    const int ab = buf * (BM * LDK);
#pragma unroll
    for (int g = 0; g < 2; ++g)
      async_ld_b128(&sA[ab + (rb + g * 64) * LDK + kq], aptr[g] + kb + kq);
#pragma unroll
    for (int g = 0; g < 2; ++g) {
      int n = rb + g * 64;
      async_ld_b128(&sB[ab + n * LDK + kq], wd + (size_t)n * I_DIM + kb + kq);
    }
  };

  issue(0, 0);
  wait_async0();
  __syncthreads();

  const int S = I_DIM / KB;                     // 44 stages
  for (int s = 0; s < S; ++s) {
    const int cur = s & 1;
    if (s + 1 < S) issue((s + 1) * KB, cur ^ 1);

    const bf16* A = sA + cur * (BM * LDK);
    const bf16* B = sB + cur * (BN * LDK);
    v16bf af[2], bf_[4];
#pragma unroll
    for (int mi = 0; mi < 2; ++mi) {
      int m = wm * 32 + mi * 16 + ln;
      af[mi] = cat16(*(const bf16x8*)&A[m * LDK + q * 8],
                     *(const bf16x8*)&A[m * LDK + 16 + q * 8]);
    }
#pragma unroll
    for (int ni = 0; ni < 4; ++ni) {
      int n = wn * 64 + ni * 16 + ln;
      bf_[ni] = cat16(*(const bf16x8*)&B[n * LDK + q * 8],
                      *(const bf16x8*)&B[n * LDK + 16 + q * 8]);
    }
#pragma unroll
    for (int mi = 0; mi < 2; ++mi)
#pragma unroll
      for (int ni = 0; ni < 4; ++ni)
        acc[mi][ni] = __builtin_amdgcn_wmma_f32_16x16x32_bf16(
            false, af[mi], false, bf_[ni], (short)0, acc[mi][ni], false, false);

    if (s + 1 < S) {
      wait_async0();
      __syncthreads();
    }
  }

  // epilogue: weighted scatter-add (exactly 2 commutative adds per element)
#pragma unroll
  for (int mi = 0; mi < 2; ++mi)
#pragma unroll
    for (int ni = 0; ni < 4; ++ni) {
      int col = nbase + wn * 64 + ni * 16 + ln;
#pragma unroll
      for (int r = 0; r < 8; ++r) {
        int gr = mbase + wm * 32 + mi * 16 + r + q * 8;
        if (gr < cnt) {
          int   row = off_e + gr;
          int   tok = perm_tok[row];
          float wgt = perm_w[row];
          atomicAdd(&out[(size_t)tok * H_DIM + col], acc[mi][ni][r] * wgt);
        }
      }
    }
}

// ---------------- host launch ----------------
extern "C" void kernel_launch(void* const* d_in, const int* in_sizes, int n_in,
                              void* d_out, int out_size, void* d_ws, size_t ws_size,
                              hipStream_t stream) {
  (void)in_sizes; (void)n_in; (void)out_size; (void)ws_size;
  const float* x   = (const float*)d_in[0];
  const float* rw  = (const float*)d_in[1];
  const float* wgu = (const float*)d_in[2];
  const float* wdn = (const float*)d_in[3];
  float* out = (float*)d_out;

  char* ws = (char*)d_ws;
  size_t off = 0;
  auto alloc = [&](size_t bytes) -> void* {
    void* p = ws + off;
    off = (off + bytes + 255) & ~(size_t)255;
    return p;
  };
  bf16*  xb       = (bf16*) alloc((size_t)T_TOK * H_DIM * sizeof(bf16));
  bf16*  wgut     = (bf16*) alloc((size_t)E_NUM * 2 * I_DIM * H_DIM * sizeof(bf16)); // (2I x H)
  bf16*  wdnt     = (bf16*) alloc((size_t)E_NUM * H_DIM * I_DIM * sizeof(bf16));     // (H x I)
  bf16*  hbuf     = (bf16*) alloc((size_t)T_TOK * TOPK * I_DIM * sizeof(bf16));
  int*   topk_idx = (int*)  alloc((size_t)T_TOK * TOPK * sizeof(int));
  float* topk_w   = (float*)alloc((size_t)T_TOK * TOPK * sizeof(float));
  int*   perm_tok = (int*)  alloc((size_t)T_TOK * TOPK * sizeof(int));
  float* perm_wt  = (float*)alloc((size_t)T_TOK * TOPK * sizeof(float));
  int*   counts   = (int*)  alloc(E_NUM * sizeof(int));
  int*   offsets  = (int*)  alloc((E_NUM + 1) * sizeof(int));
  int*   fill     = (int*)  alloc(E_NUM * sizeof(int));

  hipMemsetAsync(out, 0, (size_t)T_TOK * H_DIM * sizeof(float), stream);
  hipMemsetAsync(counts, 0, E_NUM * sizeof(int), stream);

  cvt_f32_bf16<<<2048, 256, 0, stream>>>(x, xb, T_TOK * H_DIM / 4);
  transpose_cvt<<<dim3((2 * I_DIM) / 64, H_DIM / 64, E_NUM), 256, 0, stream>>>(
      wgu, wgut, H_DIM, 2 * I_DIM);
  transpose_cvt<<<dim3(H_DIM / 64, I_DIM / 64, E_NUM), 256, 0, stream>>>(
      wdn, wdnt, I_DIM, H_DIM);

  router_kernel<<<T_TOK, 128, 0, stream>>>(x, rw, topk_idx, topk_w, counts);
  prefix_kernel<<<1, 32, 0, stream>>>(counts, offsets, fill);
  scatter_kernel<<<(T_TOK * TOPK + 255) / 256, 256, 0, stream>>>(
      topk_idx, topk_w, offsets, fill, perm_tok, perm_wt);

  moe_gemm1<<<dim3((T_TOK * TOPK) / BM, I_DIM / BN, E_NUM), 256, 0, stream>>>(
      xb, wgut, perm_tok, offsets, hbuf);
  moe_gemm2<<<dim3((T_TOK * TOPK) / BM, H_DIM / BN, E_NUM), 256, 0, stream>>>(
      hbuf, wdnt, perm_tok, perm_wt, offsets, out);
}
